// CosineAttention_63247688401490
// MI455X (gfx1250) — matrix operand order repaired and checked by reading, not compiled, not run on hardware
//
#include <hip/hip_runtime.h>
#include <hip/hip_bf16.h>

// ---------------- problem constants ----------------
#define Bv 2
#define Nv 2048
#define Fv 1024
#define Hv 16
#define Dv 64
#define QKVC (3 * Hv * Dv)   // 3072

// ---------------- types ----------------
typedef __bf16 bf16_t;
typedef __attribute__((ext_vector_type(16))) __bf16 v16bf;
typedef __attribute__((ext_vector_type(8)))  float  v8f;
typedef __attribute__((ext_vector_type(4)))  float  f32x4;
typedef __attribute__((ext_vector_type(4)))  unsigned int u32x4;

union OpBF {           // one WMMA bf16 operand (16 bf16 per lane, 8 VGPRs)
    v16bf v;
    u32x4 u[2];
    bf16_t e[16];
};
union Cf {             // one WMMA f32 accumulator (8 f32 per lane)
    v8f v;
    float f[8];
};

#define WMMA_BF16(a, b, c) \
    __builtin_amdgcn_wmma_f32_16x16x32_bf16(false, (a), false, (b), (short)0, (c), false, false)

// =====================================================================
// Prep: one-shot f32 -> bf16 conversion (x, w_qkv, w_out).
// =====================================================================
__global__ __launch_bounds__(256) void cvt_bf16_kernel(
    const float* __restrict__ src, bf16_t* __restrict__ dst, int n)
{
    const int i = (blockIdx.x * 256 + threadIdx.x) * 8;
    if (i >= n) return;
    f32x4 a = *(const f32x4*)(src + i);
    f32x4 b = *(const f32x4*)(src + i + 4);
    bf16_t tmp[8];
#pragma unroll
    for (int k = 0; k < 4; ++k) { tmp[k] = (bf16_t)a[k]; tmp[4 + k] = (bf16_t)b[k]; }
    *(u32x4*)(dst + i) = *(const u32x4*)tmp;
}

// =====================================================================
// Kernel A: QKV projection (all-bf16 operands, double-buffered K loop).
//   C[BN x 3072] = Xb[BN x 1024] @ Wqkvb^T
// Scatter: col = h*192 + d*3 + s  (qkv innermost).
//   s==0 -> q bf16 (B,H,N,D)
//   s==1 -> kv[0] f32 + k bf16 (B,H,N,D)
//   s==2 -> kv[1] f32 + v^T bf16 (B,H,D,N)
// =====================================================================
__global__ __launch_bounds__(256) void qkv_proj_kernel(
    const bf16_t* __restrict__ xb, const bf16_t* __restrict__ wqb,
    bf16_t* __restrict__ qb, bf16_t* __restrict__ kb, bf16_t* __restrict__ vtb,
    float* __restrict__ kvout)
{
    const int lane = threadIdx.x & 31;
    const int wv   = threadIdx.x >> 5;                 // 0..7
    const int r0   = blockIdx.x * 16;                  // row tile over B*N
    const int c0   = (blockIdx.y * 8 + wv) * 64;       // col tile over 3072
    const int hl   = lane >> 4;                        // half-wave select
    const int l16  = lane & 15;

    const bf16_t* xrowB = xb + (size_t)(r0 + l16) * Fv + hl * 8;
    const bf16_t* wrowB = wqb + (size_t)(c0 + l16) * Fv + hl * 8;

    auto loadSet = [&](int k0, OpBF& a, OpBF b[4]) {
        a.u[0] = *(const u32x4*)(xrowB + k0);
        a.u[1] = *(const u32x4*)(xrowB + k0 + 16);
#pragma unroll
        for (int t = 0; t < 4; ++t) {
            const bf16_t* wrow = wrowB + (size_t)t * 16 * Fv + k0;
            b[t].u[0] = *(const u32x4*)(wrow);
            b[t].u[1] = *(const u32x4*)(wrow + 16);
        }
    };

    Cf c[4] = {};
    OpBF a0, a1, b0[4], b1[4];
    loadSet(0, a0, b0);
    for (int k0 = 0; k0 < Fv; k0 += 64) {
        loadSet(k0 + 32, a1, b1);                    // prefetch next half-chunk
#pragma unroll
        for (int t = 0; t < 4; ++t) c[t].v = WMMA_BF16(a0.v, b0[t].v, c[t].v);
        const int kn = (k0 + 64) & (Fv - 1);         // wrap: no tail branch
        loadSet(kn, a0, b0);
#pragma unroll
        for (int t = 0; t < 4; ++t) c[t].v = WMMA_BF16(a1.v, b1[t].v, c[t].v);
    }

    // scatter results
#pragma unroll
    for (int t = 0; t < 4; ++t) {
#pragma unroll
        for (int r = 0; r < 8; ++r) {
            const int m   = r + hl * 8;
            const int row = r0 + m;
            const int b   = row >> 11;            // /2048
            const int nn  = row & 2047;
            const int col = c0 + t * 16 + l16;
            const int h   = col / 192;
            const int rem = col % 192;
            const int d   = rem / 3;
            const int s   = rem % 3;
            const float val = c[t].f[r];
            const size_t bhnd = (((size_t)b * Hv + h) * Nv + nn) * Dv + d;
            if (s == 0) {
                qb[bhnd] = (bf16_t)val;
            } else if (s == 1) {
                kvout[bhnd] = val;                 // kv[0] = k
                kb[bhnd]    = (bf16_t)val;
            } else {
                kvout[(size_t)Bv * Hv * Nv * Dv + bhnd] = val;   // kv[1] = v
                vtb[(((size_t)b * Hv + h) * Dv + d) * Nv + nn] = (bf16_t)val;
            }
        }
    }
}

// =====================================================================
// Kernel B: talking-heads attention + fused output projection.
// 16 waves (= 16 heads) per workgroup, one 16-row query tile per block.
// Two-pass online softmax (post-softmax head mix couples all heads).
// dotsBuf layout: [head][lane][16 regs] -> all head-mix LDS traffic is b128.
// =====================================================================
__global__ __launch_bounds__(512) void talking_heads_attn_kernel(
    const bf16_t* __restrict__ wob,
    const float* __restrict__ w_pre,  const float* __restrict__ b_pre,
    const float* __restrict__ w_post, const float* __restrict__ b_post,
    const bf16_t* __restrict__ qb, const bf16_t* __restrict__ kb,
    const bf16_t* __restrict__ vtb,
    float* __restrict__ out)
{
    __shared__ float  dotsBuf[16 * 32 * 16];   // 32KB: [head][lane][16 regs]
    __shared__ bf16_t hctx[16 * 1024];         // 32KB: per-head contexts, bf16

    const int lane = threadIdx.x & 31;
    const int g    = threadIdx.x >> 5;      // wave id == head
    const int i0   = blockIdx.x * 16;       // query-row tile
    const int b    = blockIdx.y;
    const int hl   = lane >> 4;
    const int l16  = lane & 15;

    // head-mix weights for this wave's output head
    float wpre[Hv], wpost[Hv];
#pragma unroll
    for (int h = 0; h < Hv; ++h) {
        wpre[h]  = w_pre[g * Hv + h];
        wpost[h] = w_post[g * Hv + h];
    }
    const float bpre  = b_pre[g];
    const float bpost = b_post[g];

    // Q operands held in registers for the whole kernel (2 K-chunks over D)
    OpBF aq[2];
    {
        const bf16_t* qrow = qb + (((size_t)b * Hv + g) * Nv + i0 + l16) * Dv;
#pragma unroll
        for (int cc = 0; cc < 2; ++cc) {
            const bf16_t* p = qrow + cc * 32 + hl * 16;
            aq[cc].u[0] = *(const u32x4*)(p);
            aq[cc].u[1] = *(const u32x4*)(p + 8);
        }
    }

    const bf16_t* krowBase = kb + (((size_t)b * Hv + g) * Nv) * Dv;
    float* mySlot = dotsBuf + g * 512 + lane * 16;   // this wave's publish slot

    // compute scaled QK^T for a 16x32 j-tile and publish to LDS (b128 stores)
    auto dots_to_lds = [&](int j0) {
        OpBF bk[2][2];
#pragma unroll
        for (int t = 0; t < 2; ++t) {
            const bf16_t* kr = krowBase + (size_t)(j0 + t * 16 + l16) * Dv + hl * 16;
            bk[t][0].u[0] = *(const u32x4*)(kr);
            bk[t][0].u[1] = *(const u32x4*)(kr + 8);
            bk[t][1].u[0] = *(const u32x4*)(kr + 32);
            bk[t][1].u[1] = *(const u32x4*)(kr + 40);
        }
#pragma unroll
        for (int t = 0; t < 2; ++t) {
            v8f acc = {0, 0, 0, 0, 0, 0, 0, 0};
            acc = WMMA_BF16(aq[0].v, bk[t][0].v, acc);
            acc = WMMA_BF16(aq[1].v, bk[t][1].v, acc);
            Cf cs;
#pragma unroll
            for (int r = 0; r < 8; ++r) cs.f[r] = acc[r] * 0.125f;
            *(f32x4*)(mySlot + t * 8)     = *(const f32x4*)&cs.f[0];
            *(f32x4*)(mySlot + t * 8 + 4) = *(const f32x4*)&cs.f[4];
        }
    };

    // head mix (pre or post): mixed = bias + sum_h w[h] * buf[h][lane][:]
    auto headmix = [&](const float* w, float bias, float* mixed) {
        f32x4 m0 = {bias, bias, bias, bias}, m1 = m0, m2 = m0, m3 = m0;
        const float* base = dotsBuf + lane * 16;
#pragma unroll
        for (int h = 0; h < Hv; ++h) {
            const float* p = base + h * 512;
            f32x4 d0 = *(const f32x4*)(p);
            f32x4 d1 = *(const f32x4*)(p + 4);
            f32x4 d2 = *(const f32x4*)(p + 8);
            f32x4 d3 = *(const f32x4*)(p + 12);
            const float wh = w[h];
            m0 += wh * d0; m1 += wh * d1; m2 += wh * d2; m3 += wh * d3;
        }
        *(f32x4*)(mixed)      = m0;
        *(f32x4*)(mixed + 4)  = m1;
        *(f32x4*)(mixed + 8)  = m2;
        *(f32x4*)(mixed + 12) = m3;
    };

    // ---------------- pass 1: online row max / sum-exp ----------------
    float m_run[8], l_run[8];
#pragma unroll
    for (int r = 0; r < 8; ++r) { m_run[r] = -1e30f; l_run[r] = 0.0f; }

    for (int j0 = 0; j0 < Nv; j0 += 32) {
        __builtin_prefetch(krowBase + (size_t)(j0 + 32 + l16) * Dv, 0, 1);
        dots_to_lds(j0);
        __syncthreads();
        float mixed[16];
        headmix(wpre, bpre, mixed);
        __syncthreads();
#pragma unroll
        for (int r = 0; r < 8; ++r) {
            float v0 = mixed[r], v1 = mixed[8 + r];
            float tmax = fmaxf(v0, v1);
            for (int msk = 8; msk >= 1; msk >>= 1)
                tmax = fmaxf(tmax, __shfl_xor(tmax, msk, 32));
            const float mnew = fmaxf(m_run[r], tmax);
            float s = __expf(v0 - mnew) + __expf(v1 - mnew);
            for (int msk = 8; msk >= 1; msk >>= 1)
                s += __shfl_xor(s, msk, 32);
            l_run[r] = l_run[r] * __expf(m_run[r] - mnew) + s;
            m_run[r] = mnew;
        }
    }
    float rl[8];
#pragma unroll
    for (int r = 0; r < 8; ++r) rl[r] = 1.0f / l_run[r];

    // ---------------- pass 2: normalize, post-mix, attn @ V ----------------
    Cf accO[4] = {};
    for (int j0 = 0; j0 < Nv; j0 += 32) {
        dots_to_lds(j0);
        __syncthreads();
        float mixed[16];
        headmix(wpre, bpre, mixed);
        __syncthreads();

        // normalized attention for this head -> LDS (b128 stores)
        float pbuf[16];
#pragma unroll
        for (int reg = 0; reg < 16; ++reg) {
            const int r = reg & 7;
            pbuf[reg] = __expf(mixed[reg] - m_run[r]) * rl[r];
        }
#pragma unroll
        for (int q = 0; q < 4; ++q)
            *(f32x4*)(mySlot + q * 4) = *(const f32x4*)&pbuf[q * 4];
        __syncthreads();

        // post-talking-heads mix (C-layout)
        float a2[16];
        headmix(wpost, bpost, a2);
        __syncthreads();

        // V operands early so global loads overlap the LDS transpose below
        OpBF bv[4];
#pragma unroll
        for (int dt = 0; dt < 4; ++dt) {
            const bf16_t* vrow =
                vtb + (((size_t)b * Hv + g) * Dv + dt * 16 + l16) * Nv + j0 + hl * 8;
            bv[dt].u[0] = *(const u32x4*)(vrow);
            bv[dt].u[1] = *(const u32x4*)(vrow + 16);
        }

        // transpose C-layout -> A-layout through this wave's private slice
        float* slice = dotsBuf + g * 512;          // re-purposed as [m][32]
#pragma unroll
        for (int t = 0; t < 2; ++t)
#pragma unroll
            for (int r = 0; r < 8; ++r) {
                const int m = r + hl * 8;
                const int k = t * 16 + l16;
                slice[m * 32 + k] = a2[t * 8 + r];
            }
        OpBF aA;
        {
            const float* row = slice + l16 * 32 + hl * 8;
            f32x4 r0 = *(const f32x4*)(row);
            f32x4 r1 = *(const f32x4*)(row + 4);
            f32x4 r2 = *(const f32x4*)(row + 16);
            f32x4 r3 = *(const f32x4*)(row + 20);
#pragma unroll
            for (int i = 0; i < 4; ++i) {
                aA.e[i]      = (bf16_t)r0[i];
                aA.e[4 + i]  = (bf16_t)r1[i];
                aA.e[8 + i]  = (bf16_t)r2[i];
                aA.e[12 + i] = (bf16_t)r3[i];
            }
        }

        // attn2(16x32) @ V(32x64) via 4 bf16 WMMAs (V pre-transposed)
#pragma unroll
        for (int dt = 0; dt < 4; ++dt)
            accO[dt].v = WMMA_BF16(aA.v, bv[dt].v, accO[dt].v);
    }

    // assemble per-head contexts into hctx[16][1024] (bf16)
#pragma unroll
    for (int dt = 0; dt < 4; ++dt)
#pragma unroll
        for (int r = 0; r < 8; ++r) {
            const int m   = r + hl * 8;
            const int col = g * 64 + dt * 16 + l16;
            hctx[m * 1024 + col] = (bf16_t)accO[dt].f[r];
        }
    __syncthreads();

    // fused output projection: out[16 x 1024] = hctx @ w_out^T (double-buffered)
    Cf co[4] = {};
    const int c0 = g * 64;
    const bf16_t* hrowB = &hctx[l16 * 1024 + hl * 8];
    const bf16_t* wrowB = wob + (size_t)(c0 + l16) * Fv + hl * 8;
    auto loadHW = [&](int k0, OpBF& aH, OpBF bw[4]) {
        aH.u[0] = *(const u32x4*)(hrowB + k0);       // ds_load_b128
        aH.u[1] = *(const u32x4*)(hrowB + k0 + 16);
#pragma unroll
        for (int t = 0; t < 4; ++t) {
            const bf16_t* wrow = wrowB + (size_t)t * 16 * Fv + k0;
            bw[t].u[0] = *(const u32x4*)(wrow);
            bw[t].u[1] = *(const u32x4*)(wrow + 16);
        }
    };
    OpBF aH0, aH1, bw0[4], bw1[4];
    loadHW(0, aH0, bw0);
    for (int k0 = 0; k0 < Fv; k0 += 64) {
        loadHW(k0 + 32, aH1, bw1);
#pragma unroll
        for (int t = 0; t < 4; ++t) co[t].v = WMMA_BF16(aH0.v, bw0[t].v, co[t].v);
        const int kn = (k0 + 64) & (Fv - 1);
        loadHW(kn, aH0, bw0);
#pragma unroll
        for (int t = 0; t < 4; ++t) co[t].v = WMMA_BF16(aH1.v, bw1[t].v, co[t].v);
    }
#pragma unroll
    for (int t = 0; t < 4; ++t)
#pragma unroll
        for (int r = 0; r < 8; ++r) {
            const int m = r + hl * 8;
            out[((size_t)b * Nv + i0 + m) * Fv + c0 + t * 16 + l16] = co[t].f[r];
        }
}

// =====================================================================
extern "C" void kernel_launch(void* const* d_in, const int* in_sizes, int n_in,
                              void* d_out, int out_size, void* d_ws, size_t ws_size,
                              hipStream_t stream) {
    (void)in_sizes; (void)n_in; (void)out_size; (void)ws_size;
    const float* x      = (const float*)d_in[0];
    const float* w_qkv  = (const float*)d_in[1];
    const float* w_out  = (const float*)d_in[2];
    const float* w_pre  = (const float*)d_in[3];
    const float* bp_pre = (const float*)d_in[4];
    const float* w_post = (const float*)d_in[5];
    const float* bp_post= (const float*)d_in[6];

    float* out   = (float*)d_out;
    float* kvout = out + (size_t)Bv * Nv * Fv;          // kv cache region

    const size_t bhnd  = (size_t)Bv * Hv * Nv * Dv;     // 4,194,304 elems
    const size_t nX    = (size_t)Bv * Nv * Fv;          // 4,194,304
    const size_t nWqkv = (size_t)QKVC * Fv;             // 3,145,728
    const size_t nWout = (size_t)Fv * Hv * Dv;          // 1,048,576

    bf16_t* qb  = (bf16_t*)d_ws;
    bf16_t* kb  = qb + bhnd;
    bf16_t* vtb = kb + bhnd;
    bf16_t* xb  = vtb + bhnd;
    bf16_t* wqb = xb + nX;
    bf16_t* wob = wqb + nWqkv;                          // ~40 MB total ws

    // one-shot bf16 conversions (bandwidth-trivial, L2-resident afterwards)
    cvt_bf16_kernel<<<(int)(nX    / (8 * 256)), 256, 0, stream>>>(x, xb, (int)nX);
    cvt_bf16_kernel<<<(int)(nWqkv / (8 * 256)), 256, 0, stream>>>(w_qkv, wqb, (int)nWqkv);
    cvt_bf16_kernel<<<(int)(nWout / (8 * 256)), 256, 0, stream>>>(w_out, wob, (int)nWout);

    // QKV projection: 256 row tiles x (6 blocks * 8 waves) covering 3072 cols
    qkv_proj_kernel<<<dim3((Bv * Nv) / 16, QKVC / (64 * 8)), 256, 0, stream>>>(
        xb, wqb, qb, kb, vtb, kvout);

    // attention: one workgroup per (16-row tile, batch); 16 waves = 16 heads
    talking_heads_attn_kernel<<<dim3(Nv / 16, Bv), 512, 0, stream>>>(
        wob, w_pre, bp_pre, w_post, bp_post, qb, kb, vtb, out);
}
